// NonDglSAGE_35330400977322
// MI455X (gfx1250) — matrix operand order repaired and checked by reading, not compiled, not run on hardware
//
#include <hip/hip_runtime.h>
#include <hip/hip_bf16.h>

// ---------------------------------------------------------------------------
// NonDglSAGE on MI455X (gfx1250), wave32 + WMMA bf16 + TDM async tile loads.
//
//   h0 = l2norm(relu(adj @ (x  @ W0) + b0))       // (adj@x)@W == adj@(x@W)
//   h1 = l2norm(relu(adj @ (h0 @ W1) + b1))
//   out =               adj @ (h1 @ W2) + b2
//
// k_small : P = h @ W, written transposed bf16 (B-operand feed layout).
// k_big   : adj @ P with v_wmma_f32_16x16x32_bf16; adj (fp32, 576 MB) is
//           streamed from HBM exactly once per layer (~74us floor over 3
//           layers at 23.3 TB/s -> HBM-bound) and converted to bf16 inline.
//           The bf16 P tile is fetched by the Tensor Data Mover
//           (tensor_load_to_lds, TENSORcnt) with LDS padding matching the
//           20-dword conflict-free row stride. Epilogue fuses bias/relu/
//           row-L2-norm via shfl_xor butterflies + ds_add_f32 partials.
// ---------------------------------------------------------------------------

typedef __bf16        v16bf  __attribute__((ext_vector_type(16)));
typedef float         v8f    __attribute__((ext_vector_type(8)));
typedef unsigned int  u32x4  __attribute__((ext_vector_type(4)));
typedef float         f32x4  __attribute__((ext_vector_type(4)));
typedef unsigned int  su32x4 __attribute__((ext_vector_type(4)));
typedef int           si32x4 __attribute__((ext_vector_type(4)));
typedef int           si32x8 __attribute__((ext_vector_type(8)));

#define NROWS 12000
#define FEATS 256

#if defined(__has_builtin)
#if __has_builtin(__builtin_amdgcn_tensor_load_to_lds) && \
    __has_builtin(__builtin_amdgcn_s_wait_tensorcnt)
#define HAVE_TDM 1
#endif
#endif
#ifndef HAVE_TDM
#define HAVE_TDM 0
#endif

// fp32 -> bf16 pair, round-to-nearest-even, packed into one dword.
__device__ __forceinline__ unsigned bf16pk(float a, float b) {
  unsigned ua = __float_as_uint(a);
  unsigned ub = __float_as_uint(b);
  ua = ua + 0x7fffu + ((ua >> 16) & 1u);
  ub = ub + 0x7fffu + ((ub >> 16) & 1u);
  return (ua >> 16) | (ub & 0xffff0000u);
}

#if HAVE_TDM
// 2D TDM load: tile_d1 rows x tile_d0 elements (data_size=2B) from a tensor
// with row stride d0_stride (elements), into LDS at lds_addr with 16B pad
// every 64B row (pad_interval=3 -> 16 dwords, pad_amount=3 -> 4 dwords)
// => effective LDS row stride of 20 dwords. D# packing per CDNA5 ISA §8.3-8.4.
__device__ __forceinline__ void tdm_load_bf16_2d(unsigned lds_addr,
                                                 const void* gptr,
                                                 unsigned tensor_d0,
                                                 unsigned tensor_d1,
                                                 unsigned tile_d0,
                                                 unsigned tile_d1,
                                                 unsigned d0_stride) {
  unsigned long long ga = (unsigned long long)gptr;
  su32x4 g0;
  g0[0] = 1u;                                  // count=1, user descriptor
  g0[1] = lds_addr;                            // LDS byte address
  g0[2] = (unsigned)ga;                        // global_addr[31:0]
  g0[3] = (unsigned)(ga >> 32) | (2u << 30);   // global_addr[56:32] | type=2
  si32x8 g1;
  g1[0] = (int)((1u << 16)        // data_size = 1 -> 2 bytes
                | (1u << 20)      // pad_enable
                | (3u << 22)      // pad_interval: 16 dwords (64B) per burst
                | (3u << 25));    // pad_amount: 4 dwords (16B)
  g1[1] = (int)((tensor_d0 & 0xffffu) << 16);                 // tensor_dim0 lo
  g1[2] = (int)((tensor_d0 >> 16) | ((tensor_d1 & 0xffffu) << 16));
  g1[3] = (int)((tensor_d1 >> 16) | ((tile_d0 & 0xffffu) << 16));
  g1[4] = (int)(tile_d1 & 0xffffu);            // tile_dim1 (tile_dim2 = 0)
  g1[5] = (int)d0_stride;                      // tensor_dim0_stride lo
  g1[6] = 0;
  g1[7] = 0;
  si32x4 gz = {0, 0, 0, 0};
#if __has_include(<hip/amd_detail/amd_gfx1250_TDM.h>)
  si32x8 gz8 = {0, 0, 0, 0, 0, 0, 0, 0};
  __builtin_amdgcn_tensor_load_to_lds(g0, g1, gz, gz, gz8, 0);
#else
  __builtin_amdgcn_tensor_load_to_lds(g0, g1, gz, gz, 0);
#endif
}
#endif  // HAVE_TDM

// ---------------------------------------------------------------------------
// k_small: P = h @ W  (K=256).  Output written TRANSPOSED as bf16:
//   Pt[n * 12000 + r] = bf16(P[r][n]); rows [40,64) zero-padded for layer 2.
// ---------------------------------------------------------------------------
__global__ __launch_bounds__(256)
void k_small(const float* __restrict__ h, const float* __restrict__ W,
             unsigned short* __restrict__ Pt, int fanout, int nout) {
  constexpr int ROWS = 32;
  __shared__ float hsT[FEATS][ROWS];          // hsT[k][r]
  const int r0 = blockIdx.x * ROWS;

  for (int c = threadIdx.x; c < ROWS * (FEATS / 4); c += 256) {
    int r  = c / (FEATS / 4);
    int k4 = (c % (FEATS / 4)) * 4;
    f32x4 v = *(const f32x4*)(h + (size_t)(r0 + r) * FEATS + k4);
    hsT[k4 + 0][r] = v.x;
    hsT[k4 + 1][r] = v.y;
    hsT[k4 + 2][r] = v.z;
    hsT[k4 + 3][r] = v.w;
  }
  __syncthreads();

  const int n = threadIdx.x;
  float acc[ROWS];
#pragma unroll
  for (int r = 0; r < ROWS; ++r) acc[r] = 0.0f;

  for (int k = 0; k < FEATS; ++k) {
    float w = (n < fanout) ? W[(size_t)k * fanout + n] : 0.0f;
#pragma unroll
    for (int r4 = 0; r4 < ROWS; r4 += 4) {
      f32x4 hv = *(const f32x4*)&hsT[k][r4];
      acc[r4 + 0] += hv.x * w;
      acc[r4 + 1] += hv.y * w;
      acc[r4 + 2] += hv.z * w;
      acc[r4 + 3] += hv.w * w;
    }
  }

  if (n < nout) {
    unsigned pkd[ROWS / 2];
#pragma unroll
    for (int r = 0; r < ROWS; r += 2) pkd[r >> 1] = bf16pk(acc[r], acc[r + 1]);
    u32x4* dst = (u32x4*)(Pt + (size_t)n * NROWS + r0);
#pragma unroll
    for (int q = 0; q < ROWS / 8; ++q) {
      u32x4 v = {pkd[q * 4 + 0], pkd[q * 4 + 1], pkd[q * 4 + 2], pkd[q * 4 + 3]};
      dst[q] = v;
    }
  }
}

// ---------------------------------------------------------------------------
// k_big: agg = adj(12000x12000 fp32) @ P(12000xBN via PT bf16), WMMA bf16.
// Block tile 128 x BN (BN = WCOLS*64), 4 x WCOLS waves, wave tile 32x64
// (2x4 fragments), K-step 32. LDS rows padded to 20 dwords -> conflict-free
// two-b128 fragment gathers per the ISA 16-bit A/B lane layouts.
// B tile arrives via the Tensor Data Mover (wave 0 issues, TENSORcnt wait,
// workgroup barrier publishes); A tile is fp32->bf16 converted inline.
// ---------------------------------------------------------------------------
template <int WCOLS, bool FUSE>
__global__ __launch_bounds__(WCOLS * 128)
void k_big(const float* __restrict__ adj, const unsigned short* __restrict__ Pt,
           const float* __restrict__ bias, float* __restrict__ out, int ncols) {
  constexpr int BM = 128, BK = 32;
  constexpr int BN = WCOLS * 64;
  constexpr int T  = WCOLS * 128;
  constexpr int LSTR = 20;                       // dword stride per 32-bf16 row
  __shared__ unsigned sA[BM * LSTR];
  __shared__ unsigned sB[BN * LSTR];
  __shared__ float    rowsq[BM];

  const int tid  = threadIdx.x;
  const int lane = tid & 31;
  const int l15  = lane & 15;
  const int hi   = lane >> 4;                    // half-wave select
  const int wv   = tid >> 5;
  const int wr   = wv & 3;                       // wave row (4)
  const int wc   = wv >> 2;                      // wave col (WCOLS)
  const int m0   = blockIdx.x * BM;

  if (FUSE && tid < BM) rowsq[tid] = 0.0f;

  v8f acc[2][4];
  const v8f vz = {0.f, 0.f, 0.f, 0.f, 0.f, 0.f, 0.f, 0.f};
#pragma unroll
  for (int mi = 0; mi < 2; ++mi)
#pragma unroll
    for (int ni = 0; ni < 4; ++ni) acc[mi][ni] = vz;

  for (int kt = 0; kt < NROWS; kt += BK) {
    // --- P tile: BN x 32 bf16 -> LDS (padded rows) ---
#if HAVE_TDM
    if (wv == 0)
      tdm_load_bf16_2d((unsigned)(uintptr_t)sB, Pt + kt,
                       /*tensor_d0=*/NROWS, /*tensor_d1=*/BN,
                       /*tile_d0=*/BK, /*tile_d1=*/BN,
                       /*d0_stride=*/NROWS);
#else
    for (int c = tid; c < BN * 4; c += T) {
      int n  = c >> 2;
      int cu = (c & 3) << 2;                     // dword column (0,4,8,12)
      u32x4 u = *(const u32x4*)(Pt + (size_t)n * NROWS + kt + (cu << 1));
      *(u32x4*)&sB[n * LSTR + cu] = u;
    }
#endif
    // --- adj tile: 128 x 32 fp32 -> bf16 -> LDS ---
    for (int c = tid; c < BM * 8; c += T) {
      int row = c >> 3;
      int cc  = (c & 7) << 2;                    // f32 column in tile
      int gr  = m0 + row;
      f32x4 f = {0.f, 0.f, 0.f, 0.f};
      if (gr < NROWS) {
        f = *(const f32x4*)(adj + (size_t)gr * NROWS + kt + cc);
        if (kt + BK < NROWS)
          __builtin_prefetch(adj + (size_t)gr * NROWS + kt + BK + cc, 0, 0);
      }
      sA[row * LSTR + (cc >> 1) + 0] = bf16pk(f.x, f.y);
      sA[row * LSTR + (cc >> 1) + 1] = bf16pk(f.z, f.w);
    }
#if HAVE_TDM
    if (wv == 0) __builtin_amdgcn_s_wait_tensorcnt(0);
#endif
    __syncthreads();

    union Frag { u32x4 u[2]; v16bf v; } af[2], bfg[4];
#pragma unroll
    for (int mi = 0; mi < 2; ++mi) {
      int base = (wr * 32 + mi * 16 + l15) * LSTR + hi * 4;
      af[mi].u[0] = *(const u32x4*)&sA[base];
      af[mi].u[1] = *(const u32x4*)&sA[base + 8];
    }
#pragma unroll
    for (int ni = 0; ni < 4; ++ni) {
      int base = (wc * 64 + ni * 16 + l15) * LSTR + hi * 4;
      bfg[ni].u[0] = *(const u32x4*)&sB[base];
      bfg[ni].u[1] = *(const u32x4*)&sB[base + 8];
    }
#pragma unroll
    for (int mi = 0; mi < 2; ++mi)
#pragma unroll
      for (int ni = 0; ni < 4; ++ni)
        acc[mi][ni] = __builtin_amdgcn_wmma_f32_16x16x32_bf16(
            false, af[mi].v, false, bfg[ni].v, (short)0, acc[mi][ni],
            false, false);
    __syncthreads();
  }

  // per-lane bias for its 4 fragment columns
  float bl[4];
#pragma unroll
  for (int ni = 0; ni < 4; ++ni) {
    int colg = wc * 64 + ni * 16 + l15;
    bl[ni] = (colg < ncols) ? bias[colg] : 0.0f;
  }

  if (FUSE) {
    // relu(agg + b), accumulate per-row sum-of-squares across lanes + waves
#pragma unroll
    for (int mi = 0; mi < 2; ++mi)
#pragma unroll
      for (int r = 0; r < 8; ++r) {
        float s = 0.0f;
#pragma unroll
        for (int ni = 0; ni < 4; ++ni) {
          float v = acc[mi][ni][r] + bl[ni];
          v = fmaxf(v, 0.0f);
          acc[mi][ni][r] = v;
          s += v * v;
        }
        s += __shfl_xor(s, 1);
        s += __shfl_xor(s, 2);
        s += __shfl_xor(s, 4);
        s += __shfl_xor(s, 8);
        if (l15 == 0)
          atomicAdd(&rowsq[wr * 32 + mi * 16 + r + hi * 8], s);  // ds_add_f32
      }
    __syncthreads();
#pragma unroll
    for (int mi = 0; mi < 2; ++mi)
#pragma unroll
      for (int r = 0; r < 8; ++r) {
        int row  = wr * 32 + mi * 16 + r + hi * 8;
        int rowg = m0 + row;
        if (rowg >= NROWS) continue;
        float inv = 1.0f / fmaxf(sqrtf(rowsq[row]), 1e-12f);
#pragma unroll
        for (int ni = 0; ni < 4; ++ni)
          out[(size_t)rowg * ncols + wc * 64 + ni * 16 + l15] =
              acc[mi][ni][r] * inv;
      }
  } else {
#pragma unroll
    for (int mi = 0; mi < 2; ++mi)
#pragma unroll
      for (int r = 0; r < 8; ++r) {
        int rowg = m0 + wr * 32 + mi * 16 + r + hi * 8;
        if (rowg >= NROWS) continue;
#pragma unroll
        for (int ni = 0; ni < 4; ++ni) {
          int colg = wc * 64 + ni * 16 + l15;
          if (colg < ncols)
            out[(size_t)rowg * ncols + colg] = acc[mi][ni][r] + bl[ni];
        }
      }
  }
}

// ---------------------------------------------------------------------------
// Launch. Inputs: adj, x, W0, b0, W1, b1, W2, b2 (all fp32).
// Workspace: PT bf16 (6.144 MB) + h fp32 (12.288 MB) = ~18.5 MB.
// ---------------------------------------------------------------------------
extern "C" void kernel_launch(void* const* d_in, const int* in_sizes, int n_in,
                              void* d_out, int out_size, void* d_ws,
                              size_t ws_size, hipStream_t stream) {
  const float* adj = (const float*)d_in[0];
  const float* x   = (const float*)d_in[1];
  const float* W0  = (const float*)d_in[2];
  const float* b0  = (const float*)d_in[3];
  const float* W1  = (const float*)d_in[4];
  const float* b1  = (const float*)d_in[5];
  const float* W2  = (const float*)d_in[6];
  const float* b2  = (const float*)d_in[7];
  float* out = (float*)d_out;

  unsigned short* Pt = (unsigned short*)d_ws;                     // bf16 P^T
  float* hbuf = (float*)((char*)d_ws + (size_t)FEATS * NROWS * 2);

  const int gsS = NROWS / 32;            // 375
  const int gsB = (NROWS + 127) / 128;   // 94

  // layer 0
  k_small<<<gsS, 256, 0, stream>>>(x, W0, Pt, 256, 256);
  k_big<4, true><<<gsB, 512, 0, stream>>>(adj, Pt, b0, hbuf, 256);
  // layer 1
  k_small<<<gsS, 256, 0, stream>>>(hbuf, W1, Pt, 256, 256);
  k_big<4, true><<<gsB, 512, 0, stream>>>(adj, Pt, b1, hbuf, 256);
  // output layer (N=40 padded to 64 in PT)
  k_small<<<gsS, 256, 0, stream>>>(hbuf, W2, Pt, 40, 64);
  k_big<1, false><<<gsB, 128, 0, stream>>>(adj, Pt, b2, out, 40);
}
